// RniMaxPoolGCN_48790828483060
// MI455X (gfx1250) — compile-verified, matrix operand order checked by reading, not run on hardware
//
#include <hip/hip_runtime.h>

#define NN   50000
#define EE   800000
#define INC  128
#define RNIC 16
#define HC   128
#define OUTC 64
#define NPAD 50048   // 391 * 128, covers grid exactly

typedef __bf16 bf16t;
typedef __attribute__((ext_vector_type(16))) bf16t v16bf;
typedef __attribute__((ext_vector_type(8)))  float v8f;

union FragU { uint4 u[2]; v16bf v; };

__device__ __forceinline__ unsigned short f2bf(float f) {
  unsigned int u = __float_as_uint(f);
  u += 0x7FFFu + ((u >> 16) & 1u);           // round to nearest even
  return (unsigned short)(u >> 16);
}
__device__ __forceinline__ unsigned int fkey(float f) {  // order-preserving map
  unsigned int u = __float_as_uint(f);
  return (u & 0x80000000u) ? ~u : (u | 0x80000000u);
}
__device__ __forceinline__ float fdec(unsigned int k) {
  unsigned int u = (k & 0x80000000u) ? (k & 0x7FFFFFFFu) : ~k;
  return __uint_as_float(u);
}

__device__ __forceinline__ void loadA(FragU& f, const unsigned short* p) {
  f.u[0] = *(const uint4*)(p);
  f.u[1] = *(const uint4*)(p + 16);
}
__device__ __forceinline__ void loadB4(FragU b[4], const unsigned short* bsp,
                                       int lstr, int ks) {
#pragma unroll
  for (int t = 0; t < 4; ++t) {
    const unsigned short* q = bsp + t * 16 * lstr + (ks << 5);
    b[t].u[0] = *(const uint4*)(q);
    b[t].u[1] = *(const uint4*)(q + 16);
  }
}
__device__ __forceinline__ void mma4(v8f acc[4], const FragU& a, const FragU b[4]) {
#pragma unroll
  for (int t = 0; t < 4; ++t) {
    acc[t] = __builtin_amdgcn_wmma_f32_16x16x32_bf16(
        false, a.v, false, b[t].v, (short)0, acc[t], false, false);
  }
}

// ---------------------------------------------------------------------------
// bf16 WMMA GEMM: C[m][n] = sum_k A[m][k] * Bt[n][k]  (+bias, relu)
// A: bf16 [NPAD][lda], Bt: bf16 [ncols][lda]. Block = 8 waves = 128 rows x 64
// cols. B tile staged once in LDS (conflict-free padded stride). A stream is
// double-buffered via scalar fragments in a manually 2x-unrolled k-loop
// (no dynamic register-array indexing -> no scratch spill, no WAR copies).
// ---------------------------------------------------------------------------
#define LDS_PAD 8
__global__ __launch_bounds__(256) void wmma_gemm(
    const unsigned short* __restrict__ A, int lda,
    const unsigned short* __restrict__ Bt,
    float* __restrict__ Cf, unsigned short* __restrict__ Cbf, int ldc,
    const float* __restrict__ bias, int nrows, int relu)
{
  __shared__ unsigned short Bs[64 * (256 + LDS_PAD)];   // max 33 KB

  const int tid   = threadIdx.x;
  const int wave  = tid >> 5;
  const int lane  = tid & 31;
  const int l15   = lane & 15;
  const int koff8 = (lane >> 4) << 3;           // +8 for upper half-wave
  const int mrow  = blockIdx.x * 128 + wave * 16;
  const int ncol  = blockIdx.y * 64;
  const int lstr  = lda + LDS_PAD;              // LDS row stride (ushorts)

  // ---- cooperative B tile load: 64 rows x lda, 16B chunks ----
  const int cpr = lda >> 3;                     // 16B chunks per row
  const int nchunk = cpr << 6;                  // 64 rows
  for (int cc = tid; cc < nchunk; cc += 256) {
    int n = cc / cpr, j = cc - n * cpr;
    *(uint4*)(Bs + n * lstr + (j << 3)) =
        *(const uint4*)(Bt + (size_t)(ncol + n) * lda + (j << 3));
  }
  __syncthreads();

  v8f acc[4] = {};
  const int ksteps = lda >> 5;
  const unsigned short* ap = A + (size_t)(mrow + l15) * lda + koff8;
  const unsigned short* bsp = Bs + l15 * lstr + koff8;

  FragU a0, a1, b0[4], b1[4];
  loadA(a0, ap);

  int ks = 0;
  while (ks + 2 <= ksteps) {                    // two k-steps per iteration
    loadA(a1, ap + 32);                         // A for step ks+1 in flight
    loadB4(b0, bsp, lstr, ks);
    mma4(acc, a0, b0);                          // 4 back-to-back WMMAs
    if (ks + 2 < ksteps) loadA(a0, ap + 64);    // A for step ks+2 in flight
    loadB4(b1, bsp, lstr, ks + 1);
    mma4(acc, a1, b1);
    ap += 64;
    ks += 2;
  }
  if (ks < ksteps) {                            // odd remainder (ksteps == 5)
    loadB4(b0, bsp, lstr, ks);
    mma4(acc, a0, b0);
  }

#pragma unroll
  for (int t = 0; t < 4; ++t) {
    const int col = ncol + t * 16 + l15;
    const float bb = bias ? bias[col] : 0.0f;
#pragma unroll
    for (int v = 0; v < 8; ++v) {
      const int m = mrow + koff8 + v;
      float val = acc[t][v] + bb;
      if (relu) val = fmaxf(val, 0.0f);
      const size_t off = (size_t)m * ldc + col;
      if (Cbf) {
        Cbf[off] = (m < nrows) ? f2bf(val) : (unsigned short)0;
      } else if (m < nrows) {
        Cf[off] = val;
      }
    }
  }
}

// --- weight transpose + fp32->bf16, zero-padded K ---------------------------
__global__ void prep_weight(const float* __restrict__ W, unsigned short* __restrict__ Wt,
                            int K, int Hc, int Kpad) {
  int idx = blockIdx.x * blockDim.x + threadIdx.x;
  if (idx >= Hc * Kpad) return;
  int n = idx / Kpad, k = idx - n * Kpad;
  Wt[(size_t)n * Kpad + k] = (k < K) ? f2bf(W[(size_t)k * Hc + n]) : (unsigned short)0;
}

// --- degree: dinv = rsqrt(1 + indeg) ----------------------------------------
__global__ void fill_ones(float* d, int n) {
  int i = blockIdx.x * blockDim.x + threadIdx.x;
  if (i < n) d[i] = 1.0f;
}
__global__ void edge_deg(const int* __restrict__ dst, float* __restrict__ deg, int e) {
  int i = blockIdx.x * blockDim.x + threadIdx.x;
  if (i < e) atomicAdd(&deg[dst[i]], 1.0f);
}
__global__ void rsqrt_k(float* d, int n) {
  int i = blockIdx.x * blockDim.x + threadIdx.x;
  if (i < n) d[i] = rsqrtf(d[i]);
}

// --- layer-0 A: [x | noise | 0pad] as bf16, pad rows zeroed -----------------
__global__ void build_a0(const float* __restrict__ x, const float* __restrict__ noise,
                         unsigned short* __restrict__ A) {
  long idx = (long)blockIdx.x * blockDim.x + threadIdx.x;
  if (idx >= (long)NPAD * 160) return;
  int i = (int)(idx / 160), k = (int)(idx - (long)i * 160);
  unsigned short v = 0;
  if (i < NN) {
    if (k < INC)              v = f2bf(x[(size_t)i * INC + k]);
    else if (k < INC + RNIC)  v = f2bf(noise[(size_t)i * RNIC + (k - INC)]);
  }
  A[idx] = v;
}

// --- zero accumulator + reset max-pool keys ---------------------------------
__global__ void zero_agg(float* __restrict__ agg, unsigned int* __restrict__ mpk) {
  long idx = (long)blockIdx.x * blockDim.x + threadIdx.x;
  if (idx < (long)NN * HC) agg[idx] = 0.0f;
  if (idx < HC) mpk[idx] = 0u;   // 0 is below every valid fkey
}

// --- symmetric-normalized scatter-add over edges (4 channels / thread) ------
__global__ void edge_scatter(const int* __restrict__ src, const int* __restrict__ dst,
                             const float* __restrict__ dinv, const float* __restrict__ xw,
                             float* __restrict__ agg) {
  long idx = (long)blockIdx.x * blockDim.x + threadIdx.x;
  if (idx >= (long)EE * 32) return;
  int e  = (int)(idx >> 5);
  int c4 = (int)(idx & 31) << 2;
  int s = src[e], d = dst[e];
  float nv = dinv[s] * dinv[d];
  const float4 v = *(const float4*)(xw + (size_t)s * HC + c4);
  float* a = agg + (size_t)d * HC + c4;
  atomicAdd(a + 0, v.x * nv);
  atomicAdd(a + 1, v.y * nv);
  atomicAdd(a + 2, v.z * nv);
  atomicAdd(a + 3, v.w * nv);
}

// --- self-loop + bias; write bf16 into merge-A cols 0..127; global max ------
__global__ void finalize_conv(const float* __restrict__ agg, const float* __restrict__ xw,
                              const float* __restrict__ dinv, const float* __restrict__ b,
                              unsigned short* __restrict__ Amerge, unsigned int* __restrict__ mpk) {
  long idx = (long)blockIdx.x * blockDim.x + threadIdx.x;
  if (idx >= (long)NN * HC) return;
  int i = (int)(idx >> 7), c = (int)(idx & 127);
  float di = dinv[i];
  float v = agg[idx] + xw[idx] * di * di + b[c];
  Amerge[(size_t)i * 256 + c] = f2bf(v);
  atomicMax(&mpk[c], fkey(v));
}

// --- merge-A cols 128..255 = broadcast max; pad rows fully zeroed -----------
__global__ void fill_merge(unsigned short* __restrict__ Amerge, const unsigned int* __restrict__ mpk) {
  long idx = (long)blockIdx.x * blockDim.x + threadIdx.x;
  if (idx >= (long)NPAD * 256) return;
  int i = (int)(idx >> 8), c = (int)(idx & 255);
  if (i >= NN)        Amerge[idx] = 0;
  else if (c >= HC)   Amerge[idx] = f2bf(fdec(mpk[c - HC]));
}

// ---------------------------------------------------------------------------
extern "C" void kernel_launch(void* const* d_in, const int* in_sizes, int n_in,
                              void* d_out, int out_size, void* d_ws, size_t ws_size,
                              hipStream_t stream) {
  const float* x     = (const float*)d_in[0];
  const int*   ei    = (const int*)d_in[1];
  const int*   src   = ei;
  const int*   dst   = ei + EE;
  const float* noise = (const float*)d_in[3];
  const float* W0 = (const float*)d_in[4];  const float* b0  = (const float*)d_in[5];
  const float* W1 = (const float*)d_in[6];  const float* b1  = (const float*)d_in[7];
  const float* W2 = (const float*)d_in[8];  const float* b2  = (const float*)d_in[9];
  const float* M0 = (const float*)d_in[10]; const float* mb0 = (const float*)d_in[11];
  const float* M1 = (const float*)d_in[12]; const float* mb1 = (const float*)d_in[13];
  const float* M2 = (const float*)d_in[14]; const float* mb2 = (const float*)d_in[15];
  float* out = (float*)d_out;
  char* ws = (char*)d_ws;

  // scratch layout (all 16B aligned for b128 fragment loads)
  unsigned short* Aconv  = (unsigned short*)(ws);              // NPAD*160 bf16
  unsigned short* Amerge = (unsigned short*)(ws + 16015360);   // NPAD*256 bf16
  float*          xw     = (float*)(ws + 41639936);            // NPAD*128 f32
  float*          agg    = (float*)(ws + 67264512);            // NN*128 f32
  float*          dinv   = (float*)(ws + 92864512);            // NN f32
  unsigned int*   mpk    = (unsigned int*)(ws + 93065216);     // 128 keys
  unsigned short* W0t = (unsigned short*)(ws + 93066240);      // 128*160
  unsigned short* W1t = W0t + 128 * 160;                       // 128*128
  unsigned short* W2t = W1t + 128 * 128;                       // 128*128
  unsigned short* M0t = W2t + 128 * 128;                       // 128*256
  unsigned short* M1t = M0t + 128 * 256;                       // 128*256
  unsigned short* M2t = M1t + 128 * 256;                       // 64*256

  auto cdiv = [](long a, long b) { return (int)((a + b - 1) / b); };

  // weight prep (transpose + bf16, K zero-padded)
  prep_weight<<<cdiv(128 * 160, 256), 256, 0, stream>>>(W0, W0t, 144, 128, 160);
  prep_weight<<<cdiv(128 * 128, 256), 256, 0, stream>>>(W1, W1t, 128, 128, 128);
  prep_weight<<<cdiv(128 * 128, 256), 256, 0, stream>>>(W2, W2t, 128, 128, 128);
  prep_weight<<<cdiv(128 * 256, 256), 256, 0, stream>>>(M0, M0t, 256, 128, 256);
  prep_weight<<<cdiv(128 * 256, 256), 256, 0, stream>>>(M1, M1t, 256, 128, 256);
  prep_weight<<<cdiv(64  * 256, 256), 256, 0, stream>>>(M2, M2t, 256, 64, 256);

  // dinv = rsqrt(1 + indegree)
  fill_ones<<<cdiv(NN, 256), 256, 0, stream>>>(dinv, NN);
  edge_deg <<<cdiv(EE, 256), 256, 0, stream>>>(dst, dinv, EE);
  rsqrt_k  <<<cdiv(NN, 256), 256, 0, stream>>>(dinv, NN);

  // layer-0 activation buffer [x | noise | pad]
  build_a0<<<cdiv((long)NPAD * 160, 256), 256, 0, stream>>>(x, noise, Aconv);

  for (int layer = 0; layer < 3; ++layer) {
    const unsigned short* Wt = (layer == 0) ? W0t : (layer == 1) ? W1t : W2t;
    const float*          bb = (layer == 0) ? b0  : (layer == 1) ? b1  : b2;
    const int lda = (layer == 0) ? 160 : 128;

    // xw = A @ W  (fp32 out)
    dim3 gconv(NPAD / 128, HC / 64);
    wmma_gemm<<<gconv, 256, 0, stream>>>(Aconv, lda, Wt, xw, nullptr, HC, nullptr, NN, 0);

    // normalized aggregation + self loop + bias + global max
    zero_agg<<<cdiv((long)NN * HC, 256), 256, 0, stream>>>(agg, mpk);
    edge_scatter<<<cdiv((long)EE * 32, 256), 256, 0, stream>>>(src, dst, dinv, xw, agg);
    finalize_conv<<<cdiv((long)NN * HC, 256), 256, 0, stream>>>(agg, xw, dinv, bb, Amerge, mpk);
    fill_merge<<<cdiv((long)NPAD * 256, 256), 256, 0, stream>>>(Amerge, mpk);

    // merge MLP: [h | maxpool] @ M (+bias, relu except last)
    const unsigned short* Mt  = (layer == 0) ? M0t : (layer == 1) ? M1t : M2t;
    const float*          mbb = (layer == 0) ? mb0 : (layer == 1) ? mb1 : mb2;
    if (layer < 2) {
      dim3 gm(NPAD / 128, HC / 64);
      wmma_gemm<<<gm, 256, 0, stream>>>(Amerge, 256, Mt, nullptr, Aconv, HC, mbb, NN, 1);
    } else {
      dim3 gm(NPAD / 128, OUTC / 64);
      wmma_gemm<<<gm, 256, 0, stream>>>(Amerge, 256, Mt, out, nullptr, OUTC, mbb, NN, 0);
    }
  }
}
